// _PVConv_5033701671434
// MI455X (gfx1250) — matrix operand order repaired and checked by reading, not compiled, not run on hardware
//
#include <hip/hip_runtime.h>
#include <hip/hip_bf16.h>
#include <stdint.h>

// ---------------------------------------------------------------------------
// PVConv block for MI455X (gfx1250): all GEMM-shaped work through
// v_wmma_f32_16x16x32_f16 (f16 inputs, f32 accumulate). Conv3d weights are
// staged in LDS with CDNA5 async loads (ASYNCcnt) and the K-loop is
// software-pipelined so WMMAs overlap the B-fragment vmem loads.
// ---------------------------------------------------------------------------

typedef __attribute__((ext_vector_type(16))) _Float16 v16h;
typedef __attribute__((ext_vector_type(8)))  float    v8f;

#define BATCH 8
#define NPTS  65536
#define CIN   32
#define COUT  64
#define RVOX  32
#define NVOX  (RVOX*RVOX*RVOX)     // 32768
#define EPSGN 1e-5f

#define W3D_BYTES (54 * 4 * 32 * 16 * 2)   // 221184 B of packed conv weights

__device__ __forceinline__ float siluf(float x) { return x / (1.0f + __expf(-x)); }

// CDNA5 async global->LDS copy (GLOBAL_LOAD_ASYNC_TO_LDS_B128, ASYNCcnt).
__device__ __forceinline__ void async_copy_b128(uint32_t lds_off, uint64_t gaddr) {
    asm volatile("global_load_async_to_lds_b128 %0, %1, off"
                 :: "v"(lds_off), "v"(gaddr) : "memory");
}
__device__ __forceinline__ void wait_async0() {
    asm volatile("s_wait_asynccnt 0" ::: "memory");
}

// ---------------------------------------------------------------------------
// Weight packing into WMMA A-fragment order.
// A-matrix 16x32 f16 packing used here (contiguous k): lane L holds row m = L&15.
// halfword h (0..15): pair=h>>1, j=h&1,
//   k = (L>=16 ? 16:0) + h
// Fragment storage: Apack[((ks*4 + mt)*32 + lane)*16 + h]
// ---------------------------------------------------------------------------
__global__ void k_pack_a(const float* __restrict__ W, _Float16* __restrict__ Apack, int K) {
    int tid = blockIdx.x * blockDim.x + threadIdx.x;
    int Ksteps = K >> 5;
    if (tid >= Ksteps * 128) return;
    int lane = tid & 31, mt = (tid >> 5) & 3, ks = tid >> 7;
    int m = mt * 16 + (lane & 15);
    _Float16* o = Apack + ((long)(ks * 4 + mt) * 32 + lane) * 16;
    #pragma unroll
    for (int h = 0; h < 16; ++h) {
        int pair = h >> 1, j = h & 1;
        int k = ks * 32 + ((lane >= 16) ? 16 : 0) + h;
        o[h] = (_Float16)W[m * K + k];
    }
}

// Pack 3x3x3 conv weights W[o][c][kd][kh][kw] -> fragments keyed by
// kstep = tap*2 + chalf (c within step = chalf*32 + pattern(k)).
__global__ void k_pack_w3d(const float* __restrict__ W, _Float16* __restrict__ Wp) {
    int tid = blockIdx.x * blockDim.x + threadIdx.x;
    if (tid >= 54 * 128) return;
    int lane = tid & 31, mt = (tid >> 5) & 3, ks = tid >> 7;   // ks in 0..53
    int tap = ks >> 1, ch = ks & 1;
    int m = mt * 16 + (lane & 15);
    _Float16* o = Wp + ((long)(ks * 4 + mt) * 32 + lane) * 16;
    #pragma unroll
    for (int h = 0; h < 16; ++h) {
        int pair = h >> 1, j = h & 1;
        int c = ch * 32 + ((lane >= 16) ? 16 : 0) + h;
        o[h] = (_Float16)W[(m * 64 + c) * 27 + tap];
    }
}

// feats [b][c][n] f32 -> channels-last f16 [b][n][32]
__global__ void k_pack_feats(const float* __restrict__ x, _Float16* __restrict__ o) {
    int tid = blockIdx.x * blockDim.x + threadIdx.x;
    if (tid >= BATCH * NPTS * CIN) return;
    int c = tid & 31;
    int n = (tid >> 5) & (NPTS - 1);
    int b = tid >> 21;
    o[tid] = (_Float16)x[((long)(b * CIN + c)) * NPTS + n];
}

// ---------------------------------------------------------------------------
// Generic channels-last WMMA GEMM: out[b][n][64] f32 = A(64xKS*32) * in[b][n][K]
// Templated on Ksteps so all fragment loads hoist into one clause before the
// WMMA burst. One wave per 16-point tile.
// ---------------------------------------------------------------------------
template <int KS>
__global__ void k_gemm_cl(const _Float16* __restrict__ in, const _Float16* __restrict__ Apack,
                          float* __restrict__ out, int Npts) {
    const int b    = blockIdx.y;
    const int wave = threadIdx.x >> 5, lane = threadIdx.x & 31;
    const int laneN = lane & 15, hi = lane >> 4;
    const long n0 = ((long)blockIdx.x * 8 + wave) * 16;
    const _Float16* inp = in + ((long)b * Npts + n0 + laneN) * (KS * 32) + hi * 16;
    const v16h* ap = (const v16h*)Apack;

    v16h afrag[KS * 4];
    #pragma unroll
    for (int i = 0; i < KS * 4; ++i) afrag[i] = ap[i * 32 + lane];
    v16h bfrag[KS];
    #pragma unroll
    for (int ks = 0; ks < KS; ++ks) bfrag[ks] = *(const v16h*)(inp + ks * 32);

    v8f acc[4] = {};
    #pragma unroll
    for (int ks = 0; ks < KS; ++ks)
        #pragma unroll
        for (int mt = 0; mt < 4; ++mt)
            acc[mt] = __builtin_amdgcn_wmma_f32_16x16x32_f16(
                false, afrag[ks * 4 + mt], false, bfrag[ks], (short)0, acc[mt], false, false);

    // D layout: lane -> n = L&15 ; VGPR v -> m = v + 8*(L>=16); channel = mt*16 + hi*8 + v
    float* o = out + ((long)b * Npts + n0 + laneN) * 64 + hi * 8;
    #pragma unroll
    for (int mt = 0; mt < 4; ++mt) *(v8f*)(o + mt * 16) = acc[mt];
}

// ---------------------------------------------------------------------------
// Implicit-GEMM 3x3x3 conv over 32^3, 64->64 ch, SAME padding.
// Packed weights (216 KB) staged once per workgroup into LDS via
// global_load_async_to_lds_b128; A-fragments then come from ds_load while the
// per-tap B-fragment (voxel row) is prefetched one k-step ahead.
// 27 taps x 2 K-halves x 4 m-tiles = 216 WMMAs per wave.
// ---------------------------------------------------------------------------
extern __shared__ _Float16 lds_w[];

__global__ void k_conv3d(const _Float16* __restrict__ vox, const _Float16* __restrict__ Wp,
                         float* __restrict__ out) {
    const int tid = threadIdx.x;

    // ---- async stage of packed weights into LDS (54 x 4KB waves of b128s) ----
    {
        uint32_t lds_base =
            (uint32_t)(uintptr_t)(__attribute__((address_space(3))) _Float16*)lds_w;
        uint64_t gbase = (uint64_t)(uintptr_t)Wp;
        for (int i = 0; i < (W3D_BYTES >> 12); ++i) {
            uint32_t off = ((uint32_t)(i * 256 + tid)) << 4;
            async_copy_b128(lds_base + off, gbase + off);
        }
        wait_async0();
        __syncthreads();
    }

    const int wave = tid >> 5, lane = tid & 31;
    const int laneN = lane & 15, hi = lane >> 4;
    const int b  = blockIdx.y;
    const int t  = blockIdx.x * 8 + wave;            // 0..2047
    const int d  = t >> 6;
    const int h  = (t >> 1) & 31;
    const int w0 = (t & 1) * 16;
    const int w  = w0 + laneN;
    const _Float16* vb = vox + (long)b * NVOX * 64;
    const v16h* ap = (const v16h*)lds_w;

    auto loadB = [&](int ks) -> v16h {
        const int tap = ks >> 1, ch = ks & 1;
        const int dd = d + tap / 9 - 1;
        const int hh = h + (tap / 3) % 3 - 1;
        const int ww = w + tap % 3 - 1;
        v16h r = {};
        if (((unsigned)dd < 32u) & ((unsigned)hh < 32u) & ((unsigned)ww < 32u))
            r = *(const v16h*)(vb + ((long)(dd * 32 + hh) * 32 + ww) * 64 + ch * 32 + hi * 16);
        return r;
    };

    v8f acc[4] = {};
    v16h bcur = loadB(0);
    #pragma unroll 2
    for (int ks = 0; ks < 54; ++ks) {
        v16h bnext = {};
        if (ks + 1 < 54) bnext = loadB(ks + 1);     // in flight during the WMMAs below
        #pragma unroll
        for (int mt = 0; mt < 4; ++mt)
            acc[mt] = __builtin_amdgcn_wmma_f32_16x16x32_f16(
                false, ap[(ks * 4 + mt) * 32 + lane], false, bcur,
                (short)0, acc[mt], false, false);
        bcur = bnext;
    }

    float* o = out + ((long)b * NVOX + ((long)d * 32 + h) * 32 + w0 + laneN) * 64 + hi * 8;
    #pragma unroll
    for (int mt = 0; mt < 4; ++mt) *(v8f*)(o + mt * 16) = acc[mt];
}

// ---------------------------------------------------------------------------
// GroupNorm stats over channels-last buffer [b][n][64]: one block per (b,g).
// ---------------------------------------------------------------------------
__global__ void k_gn_stats(const float* __restrict__ x, int Npts, float2* __restrict__ stats) {
    const int b = blockIdx.x >> 3, g = blockIdx.x & 7;
    const float* xb = x + (long)b * Npts * 64 + g * 8;
    const int total = Npts * 8;
    float s = 0.f, ss = 0.f;
    for (int i = threadIdx.x; i < total; i += 256) {
        float v = xb[(long)(i >> 3) * 64 + (i & 7)];
        s += v; ss += v * v;
    }
    __shared__ float sh[256], sh2[256];
    sh[threadIdx.x] = s; sh2[threadIdx.x] = ss;
    __syncthreads();
    for (int o = 128; o > 0; o >>= 1) {
        if (threadIdx.x < (unsigned)o) { sh[threadIdx.x] += sh[threadIdx.x + o]; sh2[threadIdx.x] += sh2[threadIdx.x + o]; }
        __syncthreads();
    }
    if (threadIdx.x == 0) {
        float m = sh[0] / total;
        float var = sh2[0] / total - m * m;
        stats[blockIdx.x] = make_float2(m, rsqrtf(var + EPSGN));
    }
}

// GN + SiLU apply, f32 or f16 output (channels-last in/out).
template <bool TO_HALF>
__global__ void k_gn_apply(const float* __restrict__ x, const float2* __restrict__ stats,
                           const float* __restrict__ gs, const float* __restrict__ gb,
                           float* __restrict__ of, _Float16* __restrict__ oh,
                           int Npts, int total) {
    int tid = blockIdx.x * blockDim.x + threadIdx.x;
    if (tid >= total) return;
    int c = tid & 63;
    int b = (tid >> 6) / Npts;
    float2 st = stats[b * 8 + (c >> 3)];
    float v = (x[tid] - st.x) * st.y * gs[c] + gb[c];
    v = siluf(v);
    if (TO_HALF) oh[tid] = (_Float16)v; else of[tid] = v;
}

// ---------------------------------------------------------------------------
// Voxelize: fused GN+SiLU on the 1x1-conv output, then scatter-add (mean later).
// ---------------------------------------------------------------------------
__global__ void k_scatter(const float* __restrict__ x, const float* __restrict__ coords,
                          const float2* __restrict__ stats,
                          const float* __restrict__ gs, const float* __restrict__ gb,
                          float* __restrict__ vsum, float* __restrict__ cnt) {
    int tid = blockIdx.x * blockDim.x + threadIdx.x;
    if (tid >= BATCH * NPTS) return;
    const int b = tid >> 16;
    const float* cp = coords + (long)tid * 3;
    int ix = min(max((int)floorf(cp[0] * (float)RVOX), 0), RVOX - 1);
    int iy = min(max((int)floorf(cp[1] * (float)RVOX), 0), RVOX - 1);
    int iz = min(max((int)floorf(cp[2] * (float)RVOX), 0), RVOX - 1);
    int flat = (ix * RVOX + iy) * RVOX + iz;
    float* vs = vsum + ((long)b * NVOX + flat) * 64;
    atomicAdd(&cnt[b * NVOX + flat], 1.0f);
    const float* xp = x + (long)tid * 64;
    #pragma unroll 8
    for (int c = 0; c < 64; ++c) {
        float2 st = stats[b * 8 + (c >> 3)];
        float v = (xp[c] - st.x) * st.y * gs[c] + gb[c];
        atomicAdd(&vs[c], siluf(v));
    }
}

// vsum/max(cnt,1) -> f16 channels-last voxel grid
__global__ void k_vox_finalize(const float* __restrict__ vsum, const float* __restrict__ cnt,
                               _Float16* __restrict__ vh) {
    int tid = blockIdx.x * blockDim.x + threadIdx.x;
    if (tid >= BATCH * NVOX * 64) return;
    float c = fmaxf(cnt[tid >> 6], 1.0f);
    vh[tid] = (_Float16)(vsum[tid] / c);
}

// ---------------------------------------------------------------------------
// Devoxelize: trilinear gather -> channels-last f16 [b][n][64]
// ---------------------------------------------------------------------------
__global__ void k_devox(const float* __restrict__ voxf, const float* __restrict__ coords,
                        _Float16* __restrict__ oh) {
    int tid = blockIdx.x * blockDim.x + threadIdx.x;
    if (tid >= BATCH * NPTS) return;
    const int b = tid >> 16;
    const float* cp = coords + (long)tid * 3;
    float px = cp[0] * (RVOX - 1), py = cp[1] * (RVOX - 1), pz = cp[2] * (RVOX - 1);
    int ix = min(max((int)floorf(px), 0), RVOX - 2);
    int iy = min(max((int)floorf(py), 0), RVOX - 2);
    int iz = min(max((int)floorf(pz), 0), RVOX - 2);
    float fx = px - ix, fy = py - iy, fz = pz - iz;
    const float* vb = voxf + (long)b * NVOX * 64;
    _Float16* op = oh + (long)tid * 64;
    #pragma unroll
    for (int chunk = 0; chunk < 4; ++chunk) {
        float a[16];
        #pragma unroll
        for (int k = 0; k < 16; ++k) a[k] = 0.f;
        #pragma unroll
        for (int corner = 0; corner < 8; ++corner) {
            int dx = (corner >> 2) & 1, dy = (corner >> 1) & 1, dz = corner & 1;
            float wgt = (dx ? fx : 1.f - fx) * (dy ? fy : 1.f - fy) * (dz ? fz : 1.f - fz);
            const float* vp = vb + ((long)((ix + dx) * 32 + iy + dy) * 32 + iz + dz) * 64 + chunk * 16;
            #pragma unroll
            for (int k = 0; k < 16; ++k) a[k] += wgt * vp[k];
        }
        #pragma unroll
        for (int k = 0; k < 16; ++k) op[chunk * 16 + k] = (_Float16)a[k];
    }
}

// Final: out[b][c][n] = silu(gn(tmpf[b][n][c])) + skip[b][n][c]
__global__ void k_final(const float* __restrict__ tmpf, const float* __restrict__ skipf,
                        const float2* __restrict__ stats,
                        const float* __restrict__ gs, const float* __restrict__ gb,
                        float* __restrict__ out) {
    int tid = blockIdx.x * blockDim.x + threadIdx.x;
    if (tid >= BATCH * COUT * NPTS) return;
    int n = tid & (NPTS - 1);
    int c = (tid >> 16) & 63;
    int b = tid >> 22;
    long cl = ((long)b * NPTS + n) * 64 + c;
    float2 st = stats[b * 8 + (c >> 3)];
    float v = (tmpf[cl] - st.x) * st.y * gs[c] + gb[c];
    out[tid] = siluf(v) + skipf[cl];
}

// ---------------------------------------------------------------------------
extern "C" void kernel_launch(void* const* d_in, const int* in_sizes, int n_in,
                              void* d_out, int out_size, void* d_ws, size_t ws_size,
                              hipStream_t stream) {
    const float* feats   = (const float*)d_in[0];
    const float* coords  = (const float*)d_in[1];
    const float* W_pi    = (const float*)d_in[2];
    const float* gn_pi_s = (const float*)d_in[3];
    const float* gn_pi_b = (const float*)d_in[4];
    const float* W_v1    = (const float*)d_in[5];
    const float* gn_v1_s = (const float*)d_in[6];
    const float* gn_v1_b = (const float*)d_in[7];
    const float* W_v2    = (const float*)d_in[8];
    const float* gn_v2_s = (const float*)d_in[9];
    const float* gn_v2_b = (const float*)d_in[10];
    const float* W_f     = (const float*)d_in[11];
    const float* gn_f_s  = (const float*)d_in[12];
    const float* gn_f_b  = (const float*)d_in[13];
    const float* W_skip  = (const float*)d_in[14];
    float* out = (float*)d_out;

    // ---- carve workspace ----
    size_t off = 0;
    auto carve = [&](size_t bytes) -> char* {
        char* p = (char*)d_ws + off;
        off += (bytes + 255) & ~(size_t)255;
        return p;
    };
    float*     vsum    = (float*)    carve((size_t)BATCH * NVOX * 64 * 4);  // reused as vox_final f32
    float*     cnt     = (float*)    carve((size_t)BATCH * NVOX * 4);
    _Float16*  feats_h = (_Float16*) carve((size_t)BATCH * NPTS * CIN * 2);
    float*     x_pi    = (float*)    carve((size_t)BATCH * NPTS * 64 * 4);  // reused as tmp_f
    _Float16*  vox_h   = (_Float16*) carve((size_t)BATCH * NVOX * 64 * 2);
    float*     convo   = (float*)    carve((size_t)BATCH * NVOX * 64 * 4);
    _Float16*  devox_h = (_Float16*) carve((size_t)BATCH * NPTS * 64 * 2);
    float*     skip_f  = (float*)    carve((size_t)BATCH * NPTS * 64 * 4);
    _Float16*  Ap_pi   = (_Float16*) carve(4 * 1 * 32 * 16 * 2);
    _Float16*  Ap_f    = (_Float16*) carve(4 * 2 * 32 * 16 * 2);
    _Float16*  Ap_sk   = (_Float16*) carve(4 * 1 * 32 * 16 * 2);
    _Float16*  Wp1     = (_Float16*) carve(W3D_BYTES);
    _Float16*  Wp2     = (_Float16*) carve(W3D_BYTES);
    float2*    st_pi   = (float2*)   carve(64 * sizeof(float2));
    float2*    st_v1   = (float2*)   carve(64 * sizeof(float2));
    float2*    st_v2   = (float2*)   carve(64 * sizeof(float2));
    float2*    st_f    = (float2*)   carve(64 * sizeof(float2));
    (void)ws_size; (void)in_sizes; (void)n_in; (void)out_size;

    // zero scatter accumulators (vsum + cnt are adjacent)
    hipMemsetAsync(vsum, 0,
                   (size_t)BATCH * NVOX * 64 * 4 + (((size_t)BATCH * NVOX * 4 + 255) & ~(size_t)255),
                   stream);

    const int TB = 256;
    // ---- weight / input packing ----
    k_pack_a   <<<1, TB, 0, stream>>>(W_pi,   Ap_pi, 32);
    k_pack_a   <<<1, TB, 0, stream>>>(W_f,    Ap_f,  64);
    k_pack_a   <<<1, TB, 0, stream>>>(W_skip, Ap_sk, 32);
    k_pack_w3d <<<27, TB, 0, stream>>>(W_v1, Wp1);
    k_pack_w3d <<<27, TB, 0, stream>>>(W_v2, Wp2);
    k_pack_feats<<<(BATCH * NPTS * CIN + TB - 1) / TB, TB, 0, stream>>>(feats, feats_h);

    // ---- stage 1: 1x1 conv (WMMA), GN stats, fused GN+SiLU+scatter ----
    k_gemm_cl<1><<<dim3(NPTS / 128, BATCH), TB, 0, stream>>>(feats_h, Ap_pi, x_pi, NPTS);
    k_gn_stats<<<64, TB, 0, stream>>>(x_pi, NPTS, st_pi);
    k_scatter<<<(BATCH * NPTS + TB - 1) / TB, TB, 0, stream>>>(x_pi, coords, st_pi, gn_pi_s, gn_pi_b, vsum, cnt);
    k_vox_finalize<<<(BATCH * NVOX * 64 + TB - 1) / TB, TB, 0, stream>>>(vsum, cnt, vox_h);

    // ---- stage 2: conv3d #1 (WMMA implicit GEMM, LDS-staged weights) ----
    k_conv3d<<<dim3(256, BATCH), TB, W3D_BYTES, stream>>>(vox_h, Wp1, convo);
    k_gn_stats<<<64, TB, 0, stream>>>(convo, NVOX, st_v1);
    k_gn_apply<true><<<(BATCH * NVOX * 64 + TB - 1) / TB, TB, 0, stream>>>(
        convo, st_v1, gn_v1_s, gn_v1_b, nullptr, vox_h, NVOX, BATCH * NVOX * 64);

    // ---- stage 3: conv3d #2 + GN + SiLU -> f32 (for trilinear precision) ----
    k_conv3d<<<dim3(256, BATCH), TB, W3D_BYTES, stream>>>(vox_h, Wp2, convo);
    k_gn_stats<<<64, TB, 0, stream>>>(convo, NVOX, st_v2);
    k_gn_apply<false><<<(BATCH * NVOX * 64 + TB - 1) / TB, TB, 0, stream>>>(
        convo, st_v2, gn_v2_s, gn_v2_b, vsum /*reuse as vox_final*/, nullptr, NVOX, BATCH * NVOX * 64);

    // ---- stage 4: devoxelize, final 1x1 convs (WMMA), epilogue ----
    k_devox<<<(BATCH * NPTS + TB - 1) / TB, TB, 0, stream>>>(vsum, coords, devox_h);
    k_gemm_cl<2><<<dim3(NPTS / 128, BATCH), TB, 0, stream>>>(devox_h, Ap_f,  x_pi /*tmp_f*/, NPTS);
    k_gn_stats<<<64, TB, 0, stream>>>(x_pi, NPTS, st_f);
    k_gemm_cl<1><<<dim3(NPTS / 128, BATCH), TB, 0, stream>>>(feats_h, Ap_sk, skip_f, NPTS);
    k_final<<<(BATCH * COUT * NPTS + TB - 1) / TB, TB, 0, stream>>>(x_pi, skip_f, st_f, gn_f_s, gn_f_b, out);
}
